// AttentionLSTM_17274358464908
// MI455X (gfx1250) — matrix operand order repaired and checked by reading, not compiled, not run on hardware
//
#include <hip/hip_runtime.h>
#include <hip/hip_bf16.h>

// ---------------------------------------------------------------------------
// AttentionLSTM for MI455X (gfx1250, wave32, WMMA 16x16x32 f16->f32)
// B=16, S=256, D=16, H=256, A=128, O=1
// d_out layout: [0..16) = output (B,1) f32 ; [16..16+16*256*256) = attn f32
// ---------------------------------------------------------------------------

typedef __attribute__((ext_vector_type(16))) _Float16 v16h;
typedef __attribute__((ext_vector_type(8)))  float    v8f;

#define BB   16
#define SS   256
#define DD   16
#define HH   256
#define AA   128
#define G4   1024          // 4*H
#define ROWS 4096          // B*S

__device__ __forceinline__ float sigm(float x) { return 1.f / (1.f + __expf(-x)); }

// ---- WMMA fragment loaders (layouts per CDNA5 ISA 7.12.2) ------------------

// A: 16x32 f16, row-major source with leading dim lda (elements).
// lane<16 : K in {0..7, 16..23}; lane>=16 : K in {8..15, 24..31}
__device__ __forceinline__ v16h load_A_frag(const _Float16* Abase, int lda, int kbase) {
  int lane = threadIdx.x & 31;
  int m    = lane & 15;
  int half = lane >> 4;
  const _Float16* r = Abase + m * lda + kbase;
  v16h out;
#pragma unroll
  for (int v = 0; v < 8; ++v) {
    int kl = (v < 4) ? (half * 8 + 2 * v) : (16 + half * 8 + 2 * (v - 4));
    out[2 * v]     = r[kl];
    out[2 * v + 1] = r[kl + 1];
  }
  return out;
}

// B: 32x16 f16 taken from weights stored row-major [N][K] (so B[k][n]=W[n][k]).
// lane<16 : col n=lane, K in {0..7, 16..23} ; lane>=16 : K in {8..15, 24..31}
__device__ __forceinline__ v16h load_B_frag(const _Float16* W, int ldk, int nbase, int kbase) {
  int lane = threadIdx.x & 31;
  int n    = lane & 15;
  int ks   = (lane >> 4) * 8;
  const _Float16* p = W + (nbase + n) * ldk + kbase + ks;
  v16h out;
#pragma unroll
  for (int v = 0; v < 16; ++v) out[v] = p[(v < 8) ? v : (v + 8)];
  return out;
}

__device__ __forceinline__ v8f wmma_f16(v16h a, v16h b, v8f c) {
  return __builtin_amdgcn_wmma_f32_16x16x32_f16(false, a, false, b, (short)0, c, false, false);
}

// ---- small prep kernels ----------------------------------------------------

// x [B][S][16] f32 -> xh [(t*16+b)][32] f16 (K padded 16->32 with zeros)
__global__ void k_prep_x(const float* __restrict__ x, _Float16* __restrict__ xh) {
  int idx = blockIdx.x * blockDim.x + threadIdx.x;      // over 4096*32
  if (idx >= ROWS * 32) return;
  int r = idx >> 5, d = idx & 31;
  int t = r >> 4, b = r & 15;
  xh[idx] = (d < DD) ? (_Float16)x[(b * SS + t) * DD + d] : (_Float16)0.f;
}

// W_ih0 [1024][16] f32 -> [1024][32] f16 zero-padded
__global__ void k_pad_wih0(const float* __restrict__ W, _Float16* __restrict__ Wh) {
  int idx = blockIdx.x * blockDim.x + threadIdx.x;      // over 1024*32
  if (idx >= G4 * 32) return;
  int r = idx >> 5, k = idx & 31;
  Wh[idx] = (k < DD) ? (_Float16)W[r * DD + k] : (_Float16)0.f;
}

__global__ void k_f32_to_f16(const float* __restrict__ s, _Float16* __restrict__ d, int n) {
  int idx = blockIdx.x * blockDim.x + threadIdx.x;
  if (idx < n) d[idx] = (_Float16)s[idx];
}

__global__ void k_bias_fold(const float* __restrict__ a, const float* __restrict__ b,
                            float* __restrict__ o, int n) {
  int idx = blockIdx.x * blockDim.x + threadIdx.x;
  if (idx < n) o[idx] = a[idx] + b[idx];
}

// ---- generic WMMA GEMM: out[rows][N] = A[rows][K](f16) * W[N][K](f16)^T + bias
// block = 256 threads (8 waves); wave -> one 16x16 tile; grid (rows/16, N/128)
__global__ void k_gemm16(const _Float16* __restrict__ A, const _Float16* __restrict__ W,
                         const float* __restrict__ bias, float* __restrict__ out,
                         int K, int N) {
  int wave  = threadIdx.x >> 5;
  int lane  = threadIdx.x & 31;
  int n     = lane & 15;
  int mbase = (lane >> 4) * 8;
  int row0  = blockIdx.x * 16;
  int nbase = blockIdx.y * 128 + wave * 16;
  float bv = bias ? bias[nbase + n] : 0.f;
  v8f acc;
#pragma unroll
  for (int v = 0; v < 8; ++v) acc[v] = bv;
  const _Float16* Ab = A + row0 * K;
  for (int kk = 0; kk < K; kk += 32) {
    v16h a = load_A_frag(Ab, K, kk);
    v16h b = load_B_frag(W, K, nbase, kk);
    acc = wmma_f16(a, b, acc);
  }
#pragma unroll
  for (int v = 0; v < 8; ++v) out[(row0 + mbase + v) * N + nbase + n] = acc[v];
}

// ---- persistent-recurrence LSTM layer --------------------------------------
// Gx   : [S][16][1024] f32 (precomputed x-contribution + folded biases)
// Whh  : [1024][256] f16  (gate order i,f,g,o along rows)
// hout : [S][16][256] f16 ; hout32 optional f32 copy
// launch: <<<1, 512, 270336>>> (16 waves); wave w owns hidden cols [16w, 16w+16)
//
// CDNA5-specific: 320KB WGP LDS lets us pin gates i,f (rows 0..511, 256KB f16)
// in LDS for the whole scan, halving per-step L2 weight traffic; g,o stream
// from L2. Next-step Gx tile is prefetched under the WMMA loop.
__global__ void k_lstm_rec(const float* __restrict__ Gx, const _Float16* __restrict__ Whh,
                           _Float16* __restrict__ hout, float* __restrict__ hout32) {
  extern __shared__ _Float16 smem[];
  _Float16* wlds = smem;                      // Whh rows [0,512) : gates i,f (256KB)
  _Float16* hbuf = smem + 512 * HH;           // h_t as f16, 16 x 256 (8KB)

  int tid   = threadIdx.x;
  int wave  = tid >> 5;
  int lane  = tid & 31;
  int n     = lane & 15;
  int mbase = (lane >> 4) * 8;
  int col   = wave * 16 + n;                  // hidden column this lane updates

  // stage i/f weights into LDS (16B vectors; 512*256 f16 total)
  for (int e = tid * 8; e < 512 * HH; e += 512 * 8)
    *(float4*)(wlds + e) = *(const float4*)(Whh + e);
  for (int i = tid; i < 16 * HH; i += 512) hbuf[i] = (_Float16)0.f;

  float c[8];
#pragma unroll
  for (int v = 0; v < 8; ++v) c[v] = 0.f;
  __syncthreads();

  for (int t = 0; t < SS; ++t) {
    const float* g = Gx + t * 16 * G4;
    v8f ai, af, ag, ao;
#pragma unroll
    for (int v = 0; v < 8; ++v) {
      int m = mbase + v;
      ai[v] = g[m * G4 + 0 * HH + col];
      af[v] = g[m * G4 + 1 * HH + col];
      ag[v] = g[m * G4 + 2 * HH + col];
      ao[v] = g[m * G4 + 3 * HH + col];
    }
    // pull next step's Gx tile (64KB, streamed once) into cache under compute
    if (t + 1 < SS)
      __builtin_prefetch((const void*)(g + 16 * G4 + tid * 32), 0, 3);
#pragma unroll
    for (int kk = 0; kk < HH; kk += 32) {
      v16h a  = load_A_frag(hbuf, HH, kk);
      v16h bi = load_B_frag(wlds, HH, 0 * HH + wave * 16, kk);   // LDS
      ai = wmma_f16(a, bi, ai);
      v16h bf = load_B_frag(wlds, HH, 1 * HH + wave * 16, kk);   // LDS
      af = wmma_f16(a, bf, af);
      v16h bg = load_B_frag(Whh, HH, 2 * HH + wave * 16, kk);    // L2 stream
      ag = wmma_f16(a, bg, ag);
      v16h bo = load_B_frag(Whh, HH, 3 * HH + wave * 16, kk);    // L2 stream
      ao = wmma_f16(a, bo, ao);
    }
    __syncthreads();                          // all reads of hbuf complete
#pragma unroll
    for (int v = 0; v < 8; ++v) {
      float cn = sigm(af[v]) * c[v] + sigm(ai[v]) * tanhf(ag[v]);
      float hn = sigm(ao[v]) * tanhf(cn);
      c[v] = cn;
      int m = mbase + v;
      hbuf[m * HH + col] = (_Float16)hn;
      hout[t * (16 * HH) + m * HH + col] = (_Float16)hn;
      if (hout32) hout32[t * (16 * HH) + m * HH + col] = hn;
    }
    __syncthreads();                          // h_{t+1} visible to all waves
  }
}

// ---- pairwise additive-attention scores + row softmax ----------------------
// wa/ua : [S][16][128] f32 ; attn -> d_out+16, layout [B][S][S]
// grid (16, 16) : b x itile ; block 256 ; thread = j
__global__ void k_attn_scores(const float* __restrict__ wa, const float* __restrict__ ua,
                              const float* __restrict__ va_w, const float* __restrict__ va_b,
                              float* __restrict__ attn) {
  __shared__ float wa_s[16 * AA];
  __shared__ float sc[16 * SS];
  __shared__ float rmax[16], rinv[16];
  int tid   = threadIdx.x;
  int b     = blockIdx.x;
  int itile = blockIdx.y * 16;

  for (int e = tid; e < 16 * AA; e += 256) {
    int i = e >> 7, a = e & (AA - 1);
    wa_s[e] = wa[((itile + i) * 16 + b) * AA + a];
  }
  __syncthreads();

  int j = tid;
  const float* urow = ua + (j * 16 + b) * AA;
  float sum[16];
#pragma unroll
  for (int i = 0; i < 16; ++i) sum[i] = 0.f;
  for (int a = 0; a < AA; ++a) {
    float u  = urow[a];
    float vv = va_w[a];
#pragma unroll
    for (int i = 0; i < 16; ++i) sum[i] += tanhf(wa_s[i * AA + a] + u) * vv;
  }
  float vb = va_b[0];
#pragma unroll
  for (int i = 0; i < 16; ++i) sc[i * SS + j] = sum[i] + vb;
  __syncthreads();

  if (tid < 16) {
    int i = tid;
    float m = -3.4e38f;
    for (int jj = 0; jj < SS; ++jj) m = fmaxf(m, sc[i * SS + jj]);
    float s = 0.f;
    for (int jj = 0; jj < SS; ++jj) s += __expf(sc[i * SS + jj] - m);
    rmax[i] = m;
    rinv[i] = 1.f / s;
  }
  __syncthreads();
#pragma unroll
  for (int i = 0; i < 16; ++i)
    attn[((b * SS) + (itile + i)) * SS + j] = __expf(sc[i * SS + j] - rmax[i]) * rinv[i];
}

// ---- last-step context + FC head -------------------------------------------
// attn = d_out+16 ; lo32 : [S][16][256] f32 ; out = d_out[0..16)
__global__ void k_finalize(const float* __restrict__ attn, const float* __restrict__ lo32,
                           const float* __restrict__ fc_w, const float* __restrict__ fc_b,
                           float* __restrict__ out) {
  __shared__ float ctx_s[16 * HH];
  int h = threadIdx.x;                         // 0..255
  for (int b = 0; b < BB; ++b) {
    const float* arow = attn + (b * SS + (SS - 1)) * SS;
    float s = 0.f;
    for (int j = 0; j < SS; ++j) s += arow[j] * lo32[(j * 16 + b) * HH + h];
    ctx_s[b * HH + h] = s;
  }
  __syncthreads();
  if (h < BB) {
    int b = h;
    float acc = fc_b[0];
    for (int k = 0; k < HH; ++k)
      acc += fc_w[k] * ctx_s[b * HH + k] + fc_w[HH + k] * lo32[((SS - 1) * 16 + b) * HH + k];
    out[b] = acc;
  }
}

// ---------------------------------------------------------------------------

extern "C" void kernel_launch(void* const* d_in, const int* in_sizes, int n_in,
                              void* d_out, int out_size, void* d_ws, size_t ws_size,
                              hipStream_t stream) {
  (void)in_sizes; (void)n_in; (void)out_size; (void)ws_size;
  const float* x     = (const float*)d_in[0];
  const float* W_ih0 = (const float*)d_in[1];
  const float* W_hh0 = (const float*)d_in[2];
  const float* b_ih0 = (const float*)d_in[3];
  const float* b_hh0 = (const float*)d_in[4];
  const float* W_ih1 = (const float*)d_in[5];
  const float* W_hh1 = (const float*)d_in[6];
  const float* b_ih1 = (const float*)d_in[7];
  const float* b_hh1 = (const float*)d_in[8];
  const float* Wa_w  = (const float*)d_in[9];
  const float* Wa_b  = (const float*)d_in[10];
  const float* Ua_w  = (const float*)d_in[11];
  const float* Ua_b  = (const float*)d_in[12];
  const float* va_w  = (const float*)d_in[13];
  const float* va_b  = (const float*)d_in[14];
  const float* fc_w  = (const float*)d_in[15];
  const float* fc_b  = (const float*)d_in[16];

  float* outp = (float*)d_out;
  float* attn = outp + 16;

  // workspace carve-up (256B aligned)
  char* ws = (char*)d_ws;
  size_t off = 0;
  auto alloc = [&](size_t bytes) { size_t r = off; off += (bytes + 255) & ~(size_t)255; return r; };
  _Float16* xh    = (_Float16*)(ws + alloc((size_t)ROWS * 32 * 2));
  _Float16* wih0h = (_Float16*)(ws + alloc((size_t)G4 * 32 * 2));
  _Float16* whh0h = (_Float16*)(ws + alloc((size_t)G4 * HH * 2));
  _Float16* wih1h = (_Float16*)(ws + alloc((size_t)G4 * HH * 2));
  _Float16* whh1h = (_Float16*)(ws + alloc((size_t)G4 * HH * 2));
  _Float16* wah   = (_Float16*)(ws + alloc((size_t)AA * HH * 2));
  _Float16* uah   = (_Float16*)(ws + alloc((size_t)AA * HH * 2));
  float*    gb0   = (float*)(ws + alloc((size_t)G4 * 4));
  float*    gb1   = (float*)(ws + alloc((size_t)G4 * 4));
  float*    gx    = (float*)(ws + alloc((size_t)ROWS * G4 * 4));   // reused for both layers
  _Float16* h1h   = (_Float16*)(ws + alloc((size_t)ROWS * HH * 2));
  _Float16* loh   = (_Float16*)(ws + alloc((size_t)ROWS * HH * 2));
  float*    lo32  = (float*)(ws + alloc((size_t)ROWS * HH * 4));
  float*    wa    = (float*)(ws + alloc((size_t)ROWS * AA * 4));
  float*    ua    = (float*)(ws + alloc((size_t)ROWS * AA * 4));

  const size_t rec_lds = (size_t)(512 * HH + 16 * HH) * sizeof(_Float16);  // 264KB

  // 1) conversions / folds
  k_prep_x  <<<(ROWS * 32 + 255) / 256, 256, 0, stream>>>(x, xh);
  k_pad_wih0<<<(G4 * 32 + 255) / 256,   256, 0, stream>>>(W_ih0, wih0h);
  k_f32_to_f16<<<(G4 * HH + 255) / 256, 256, 0, stream>>>(W_hh0, whh0h, G4 * HH);
  k_f32_to_f16<<<(G4 * HH + 255) / 256, 256, 0, stream>>>(W_ih1, wih1h, G4 * HH);
  k_f32_to_f16<<<(G4 * HH + 255) / 256, 256, 0, stream>>>(W_hh1, whh1h, G4 * HH);
  k_f32_to_f16<<<(AA * HH + 255) / 256, 256, 0, stream>>>(Wa_w, wah, AA * HH);
  k_f32_to_f16<<<(AA * HH + 255) / 256, 256, 0, stream>>>(Ua_w, uah, AA * HH);
  k_bias_fold<<<(G4 + 255) / 256, 256, 0, stream>>>(b_ih0, b_hh0, gb0, G4);
  k_bias_fold<<<(G4 + 255) / 256, 256, 0, stream>>>(b_ih1, b_hh1, gb1, G4);

  // 2) layer 0: hoisted input GEMM + recurrence
  k_gemm16<<<dim3(ROWS / 16, G4 / 128), 256, 0, stream>>>(xh, wih0h, gb0, gx, 32, G4);
  k_lstm_rec<<<1, 512, rec_lds, stream>>>(gx, whh0h, h1h, nullptr);

  // 3) layer 1
  k_gemm16<<<dim3(ROWS / 16, G4 / 128), 256, 0, stream>>>(h1h, wih1h, gb1, gx, HH, G4);
  k_lstm_rec<<<1, 512, rec_lds, stream>>>(gx, whh1h, loh, lo32);

  // 4) attention projections (WMMA GEMMs)
  k_gemm16<<<dim3(ROWS / 16, 1), 256, 0, stream>>>(loh, wah, Wa_b, wa, HH, AA);
  k_gemm16<<<dim3(ROWS / 16, 1), 256, 0, stream>>>(loh, uah, Ua_b, ua, HH, AA);

  // 5) scores + softmax -> d_out+16 ; 6) context(last) + FC -> d_out[0..16)
  k_attn_scores<<<dim3(16, 16), 256, 0, stream>>>(wa, ua, va_w, va_b, attn);
  k_finalize<<<1, 256, 0, stream>>>(attn, lo32, fc_w, fc_b, outp);
}
